// SIEGphormer_3788161155662
// MI455X (gfx1250) — matrix operand order repaired and checked by reading, not compiled, not run on hardware
//
#include <hip/hip_runtime.h>
#include <hip/hip_bf16.h>

// ---------------- problem constants ----------------
#define NN   12288      // nodes
#define FF   128        // in features
#define DD   128        // hidden
#define EE   393216     // edges
#define BB   64         // batch
#define LL   256        // nodes per sample
#define DT   256        // token dim
#define HH   8          // heads
#define NS   (BB*LL)    // 16384 samples
#define SEQ  (LL+1)     // 257

typedef __attribute__((ext_vector_type(16))) _Float16 v16h;
typedef __attribute__((ext_vector_type(8)))  _Float16 v8h;
typedef __attribute__((ext_vector_type(8)))  float    v8f;
typedef int v4i __attribute__((vector_size(16)));

// ---- async global->LDS copy path (gfx1250 ASYNCcnt engine) ----
#if defined(__has_builtin)
#if __has_builtin(__builtin_amdgcn_global_load_async_to_lds_b128)
#define USE_ASYNC_LDS 1
#endif
#endif

__device__ __forceinline__ void wait_async_all() {
#ifdef USE_ASYNC_LDS
#if __has_builtin(__builtin_amdgcn_s_wait_asynccnt)
    __builtin_amdgcn_s_wait_asynccnt(0);
#else
    asm volatile("s_wait_asynccnt 0x0" ::: "memory");
#endif
#endif
}

#ifdef USE_ASYNC_LDS
__device__ __forceinline__ void async_cp16(const _Float16* g, _Float16* l) {
    __builtin_amdgcn_global_load_async_to_lds_b128(
        (__attribute__((address_space(1))) v4i*)g,
        (__attribute__((address_space(3))) v4i*)l,
        0, 0);
}
#endif

__device__ __forceinline__ v16h cat8(v8h lo, v8h hi) {
    return __builtin_shufflevector(lo, hi, 0,1,2,3,4,5,6,7,8,9,10,11,12,13,14,15);
}

// =====================================================================
// WMMA GEMM:  C[M x N] = op(A[M x KT] * BT[N x KT]^T + bias)
//  - A  : f16 row-major (lda);  BT : f16 pre-transposed N x KT
//  - Block 256 thr = 8 waves; block tile 64(M) x 64(N);
//    wave tile 16x32 -> 2 v_wmma per k-step.
//  - Double-buffered LDS; staging via global_load_async_to_lds_b128
//    (ASYNCcnt) when available, else b128 load/store copies.
// =====================================================================
template<int KT, bool RELU, typename OT>
__global__ __launch_bounds__(256) void gemm_wmma(
    const _Float16* __restrict__ A, int lda,
    const _Float16* __restrict__ BT,
    OT* __restrict__ C, int ldc,
    const float* __restrict__ bias)
{
    __shared__ _Float16 As[2][64 * 40];   // A tile: 64 rows x 32 K (pad 40)
    __shared__ _Float16 Bs[2][64 * 40];   // B tile: 64 cols x 32 K (transposed)

    const int tid  = threadIdx.x;
    const int lane = tid & 31;
    const int wave = tid >> 5;            // 0..7
    const int mW   = (wave & 3) * 16;     // wave M offset in block tile
    const int nW   = (wave >> 2) * 32;    // wave N offset in block tile
    const int m0   = blockIdx.x * 64;
    const int n0   = blockIdx.y * 64;

    const int half = lane >> 4;           // 0 | 1
    const int l16  = lane & 15;

    // staging: each thread owns one 8-half (16B) chunk of each tile
    const int sr = tid >> 2;              // row/col 0..63
    const int sc = (tid & 3) * 8;         // K chunk 0,8,16,24

    auto stage = [&](int k0, int buf) {
#ifdef USE_ASYNC_LDS
        async_cp16(&A [(size_t)(m0 + sr) * lda + k0 + sc], &As[buf][sr * 40 + sc]);
        async_cp16(&BT[(size_t)(n0 + sr) * KT  + k0 + sc], &Bs[buf][sr * 40 + sc]);
#else
        *(v8h*)&As[buf][sr * 40 + sc] = *(const v8h*)&A [(size_t)(m0 + sr) * lda + k0 + sc];
        *(v8h*)&Bs[buf][sr * 40 + sc] = *(const v8h*)&BT[(size_t)(n0 + sr) * KT  + k0 + sc];
#endif
    };

    v8f acc0 = {};
    v8f acc1 = {};

    stage(0, 0);
    wait_async_all();
    __syncthreads();

#pragma unroll
    for (int k0 = 0; k0 < KT; k0 += 32) {
        const int cur = (k0 >> 5) & 1;
        if (k0 + 32 < KT) stage(k0 + 32, cur ^ 1);   // prefetch next tile

        // A frag (16x32): lanes 0-15 row=l16, K 0..15; lanes 16-31 K 16..31
        const _Float16* ar = &As[cur][(mW + l16) * 40];
        v16h afr = cat8(*(const v8h*)(ar + half * 16),
                        *(const v8h*)(ar + half * 16 + 8));
        // B frags (32x16): lanes 0-15 col=l16 K 0..15; lanes 16-31 K 16..31
        const _Float16* br0 = &Bs[cur][(nW + l16) * 40 + half * 16];
        const _Float16* br1 = &Bs[cur][(nW + 16 + l16) * 40 + half * 16];
        v16h bf0 = cat8(*(const v8h*)br0, *(const v8h*)(br0 + 8));
        v16h bf1 = cat8(*(const v8h*)br1, *(const v8h*)(br1 + 8));

        acc0 = __builtin_amdgcn_wmma_f32_16x16x32_f16(false, afr, false, bf0,
                                                      (short)0, acc0, false, false);
        acc1 = __builtin_amdgcn_wmma_f32_16x16x32_f16(false, afr, false, bf1,
                                                      (short)0, acc1, false, false);

        wait_async_all();     // next tile landed in LDS
        __syncthreads();      // all waves done reading cur + see next tile
    }

    // C/D layout: lane -> col N=l16 ; VGPR v -> row M = v + half*8
#pragma unroll
    for (int j = 0; j < 2; ++j) {
        const v8f acc = j ? acc1 : acc0;
        const int col = n0 + nW + j * 16 + l16;
        const float bval = bias ? bias[col] : 0.0f;
#pragma unroll
        for (int v = 0; v < 8; ++v) {
            int row = m0 + mW + v + half * 8;
            float val = acc[v] + bval;
            if (RELU) val = fmaxf(val, 0.0f);
            C[(size_t)row * ldc + col] = (OT)val;
        }
    }
}

// =====================================================================
// one-time converts: f32 -> f16, weight transpose+convert (N x K f16)
// =====================================================================
__global__ void f32tof16_kernel(const float* __restrict__ in,
                                _Float16* __restrict__ out, int n) {
    int i = blockIdx.x * blockDim.x + threadIdx.x;
    if (i < n) out[i] = (_Float16)in[i];
}

__global__ void wtrans_kernel(const float* __restrict__ W,
                              _Float16* __restrict__ WT, int K, int N) {
    int idx = blockIdx.x * blockDim.x + threadIdx.x;
    if (idx >= K * N) return;
    int k = idx / N, n = idx % N;
    WT[(size_t)n * K + k] = (_Float16)W[idx];
}

// =====================================================================
// sparse propagation + mix
// =====================================================================
__global__ void zero_kernel(float* __restrict__ p, int n) {
    int i = blockIdx.x * blockDim.x + threadIdx.x;
    if (i < n) p[i] = 0.0f;
}

__global__ void prop_kernel(const float* __restrict__ hin, float* __restrict__ hout,
                            const float* __restrict__ vals,
                            const int* __restrict__ rows, const int* __restrict__ cols)
{
    long long idx = (long long)blockIdx.x * blockDim.x + threadIdx.x;
    if (idx >= (long long)EE * 32) return;
    int e  = (int)(idx >> 5);
    int c4 = (int)(idx & 31) * 4;
    int r = rows[e], c = cols[e];
    float v = vals[e];
    const float* s = &hin[(size_t)c * DD + c4];
    float*       d = &hout[(size_t)r * DD + c4];
    atomicAdd(d + 0, v * s[0]);
    atomicAdd(d + 1, v * s[1]);
    atomicAdd(d + 2, v * s[2]);
    atomicAdd(d + 3, v * s[3]);
}

__global__ void mix_kernel(const float* __restrict__ h, const float* __restrict__ f,
                           float* __restrict__ o, int n) {
    int i = blockIdx.x * blockDim.x + threadIdx.x;
    if (i < n) o[i] = 0.5f * h[i] + 0.5f * f[i];   // ALPHA = 0.5
}

// =====================================================================
// encoder hidden layers (only encoders 2/3 have nonzero inputs)
// =====================================================================
__global__ __launch_bounds__(128) void zbuild_kernel(
    const float* __restrict__ ppr, const float* __restrict__ drnl,
    const int* __restrict__ src, const int* __restrict__ dst,
    const int* __restrict__ sample_idx, const int* __restrict__ node_idx,
    const float* __restrict__ enc_w1, const float* __restrict__ enc_b1,
    _Float16* __restrict__ Z2, _Float16* __restrict__ Z3)
{
    int i = blockIdx.x;          // sample
    int k = threadIdx.x;         // dim
    int s   = sample_idx[i];
    int nid = node_idx[i];
    float pa = ppr[(long long)src[s] * NN + nid];
    float pb = ppr[(long long)dst[s] * NN + nid];
    float dv = drnl[(size_t)s * NN + nid];
    float z2 = pa * enc_w1[2*2*DD + k] + pb * enc_w1[2*2*DD + DD + k] + enc_b1[2*DD + k];
    float z3 = dv * enc_w1[3*2*DD + k] + dv * enc_w1[3*2*DD + DD + k] + enc_b1[3*DD + k];
    Z2[(size_t)i * DD + k] = (_Float16)fmaxf(z2, 0.0f);
    Z3[(size_t)i * DD + k] = (_Float16)fmaxf(z3, 0.0f);
}

// constant-fold dead encoders 0/1 (all-zero inputs) into a bias vector
__global__ __launch_bounds__(128) void encconst_kernel(
    const float* __restrict__ enc_b1, const float* __restrict__ enc_w2,
    const float* __restrict__ enc_b2, float* __restrict__ enc01)
{
    int j = threadIdx.x;
    for (int e = 0; e < 2; ++e) {
        float acc = enc_b2[e * DD + j];
        for (int k = 0; k < DD; ++k)
            acc += fmaxf(enc_b1[e * DD + k], 0.0f) * enc_w2[(size_t)e * DD * DD + k * DD + j];
        enc01[e * DD + j] = acc;
    }
}

__global__ __launch_bounds__(128) void cvec_kernel(
    const float* __restrict__ enc01, const float* __restrict__ struct_w1,
    const float* __restrict__ struct_b1, float* __restrict__ cvec)
{
    int j = threadIdx.x;
    float acc = struct_b1[j];
    for (int k = 0; k < 2 * DD; ++k)
        acc += enc01[k] * struct_w1[(size_t)k * DD + j];
    cvec[j] = acc;
}

// =====================================================================
// token sequence assembly (f16 for WMMA A-operand)
// =====================================================================
__global__ __launch_bounds__(256) void seq_kernel(
    const _Float16* __restrict__ st, const float* __restrict__ rex,
    const int* __restrict__ node_idx, const float* __restrict__ cls_tok,
    _Float16* __restrict__ seq)
{
    int r = blockIdx.x;            // 0..B*SEQ-1
    int t = threadIdx.x;           // 0..255
    int b = r / SEQ, p = r % SEQ;
    float v;
    if (p == 0) {
        v = cls_tok[t];
    } else {
        int i = b * LL + (p - 1);
        v = (t < DD) ? (float)st[(size_t)i * DD + t]
                     : rex[(size_t)node_idx[i] * DD + (t - DD)];
    }
    seq[(size_t)r * DT + t] = (_Float16)v;
}

__global__ __launch_bounds__(256) void q3gather_kernel(
    const _Float16* __restrict__ seq, _Float16* __restrict__ q3in)
{
    int r = blockIdx.x;            // 0..191
    int b = r / 3, p = r % 3;
    q3in[(size_t)r * DT + threadIdx.x] = seq[((size_t)b * SEQ + p) * DT + threadIdx.x];
}

// =====================================================================
// softmax attention: 3 queries x 257 keys, one block per (batch, head)
// =====================================================================
__global__ __launch_bounds__(256) void attn_kernel(
    const float* __restrict__ Q3, const _Float16* __restrict__ Km,
    const _Float16* __restrict__ Vm, _Float16* __restrict__ AV)
{
    const int b = blockIdx.x, h = blockIdx.y;
    __shared__ float q[3][32];
    __shared__ float sc[3][SEQ];
    __shared__ float red[256];
    const int tid = threadIdx.x;
    const float scale = 0.17677669529663687f;   // 1/sqrt(32)

    if (tid < 96) {
        int qi = tid >> 5, d = tid & 31;
        q[qi][d] = Q3[(size_t)(b * 3 + qi) * DT + h * 32 + d];
    }
    __syncthreads();

    for (int j = tid; j < SEQ; j += 256) {
        const _Float16* kr = &Km[((size_t)b * SEQ + j) * DT + h * 32];
        float s0 = 0, s1 = 0, s2 = 0;
        for (int d = 0; d < 32; ++d) {
            float kv = (float)kr[d];
            s0 += q[0][d] * kv; s1 += q[1][d] * kv; s2 += q[2][d] * kv;
        }
        sc[0][j] = s0 * scale; sc[1][j] = s1 * scale; sc[2][j] = s2 * scale;
    }
    __syncthreads();

    for (int qi = 0; qi < 3; ++qi) {
        float m = -1e30f;
        for (int j = tid; j < SEQ; j += 256) m = fmaxf(m, sc[qi][j]);
        red[tid] = m; __syncthreads();
        for (int s = 128; s > 0; s >>= 1) {
            if (tid < s) red[tid] = fmaxf(red[tid], red[tid + s]);
            __syncthreads();
        }
        float mx = red[0]; __syncthreads();
        float lsum = 0.0f;
        for (int j = tid; j < SEQ; j += 256) {
            float e = __expf(sc[qi][j] - mx);
            sc[qi][j] = e; lsum += e;
        }
        red[tid] = lsum; __syncthreads();
        for (int s = 128; s > 0; s >>= 1) {
            if (tid < s) red[tid] += red[tid + s];
            __syncthreads();
        }
        float inv = 1.0f / red[0]; __syncthreads();
        for (int j = tid; j < SEQ; j += 256) sc[qi][j] *= inv;
        __syncthreads();
    }

    const int d = tid & 31, g = tid >> 5;
    float a0 = 0, a1 = 0, a2 = 0;
    for (int j = g; j < SEQ; j += 8) {
        float vv = (float)Vm[((size_t)b * SEQ + j) * DT + h * 32 + d];
        a0 += sc[0][j] * vv; a1 += sc[1][j] * vv; a2 += sc[2][j] * vv;
    }
    float accs[3] = {a0, a1, a2};
    for (int qi = 0; qi < 3; ++qi) {
        red[tid] = accs[qi]; __syncthreads();
        if (tid < 128) red[tid] += red[tid + 128]; __syncthreads();
        if (tid < 64)  red[tid] += red[tid + 64];  __syncthreads();
        if (tid < 32) {
            red[tid] += red[tid + 32];
            AV[(size_t)(b * 3 + qi) * DT + h * 32 + tid] = (_Float16)red[tid];
        }
        __syncthreads();
    }
}

// output = [h_a | h_b | h_cls] = rows (b*3+1, b*3+2, b*3+0) of O
__global__ __launch_bounds__(256) void outscatter_kernel(
    const float* __restrict__ O, float* __restrict__ out)
{
    int b = blockIdx.x, t = threadIdx.x;
    out[(size_t)b * 768 + t]        = O[(size_t)(b * 3 + 1) * DT + t];
    out[(size_t)b * 768 + 256 + t]  = O[(size_t)(b * 3 + 2) * DT + t];
    out[(size_t)b * 768 + 512 + t]  = O[(size_t)(b * 3 + 0) * DT + t];
}

// =====================================================================
extern "C" void kernel_launch(void* const* d_in, const int* in_sizes, int n_in,
                              void* d_out, int out_size, void* d_ws, size_t ws_size,
                              hipStream_t stream) {
    (void)in_sizes; (void)n_in; (void)out_size; (void)ws_size;
    const float* x         = (const float*)d_in[0];
    const float* proj_w    = (const float*)d_in[1];
    const float* proj_b    = (const float*)d_in[2];
    const float* adj_vals  = (const float*)d_in[3];
    const int*   adj_rows  = (const int*)  d_in[4];
    const int*   adj_cols  = (const int*)  d_in[5];
    const float* ppr       = (const float*)d_in[6];
    const float* drnl      = (const float*)d_in[7];
    const int*   srcv      = (const int*)  d_in[8];
    const int*   dstv      = (const int*)  d_in[9];
    const int*   samp_idx  = (const int*)  d_in[10];
    const int*   node_idx  = (const int*)  d_in[11];
    const float* enc_w1    = (const float*)d_in[12];
    const float* enc_b1    = (const float*)d_in[13];
    const float* enc_w2    = (const float*)d_in[14];
    const float* enc_b2    = (const float*)d_in[15];
    const float* struct_w1 = (const float*)d_in[16];
    const float* struct_b1 = (const float*)d_in[17];
    const float* struct_w2 = (const float*)d_in[18];
    const float* struct_b2 = (const float*)d_in[19];
    const float* cls_tok   = (const float*)d_in[20];
    const float* wq        = (const float*)d_in[21];
    const float* wk        = (const float*)d_in[22];
    const float* wv        = (const float*)d_in[23];
    const float* wo        = (const float*)d_in[24];

    char* base = (char*)d_ws;
    size_t off = 0;
    auto alloc = [&](size_t bytes) -> void* {
        void* p = base + off; off += (bytes + 255) & ~(size_t)255; return p;
    };

    const size_t ND = (size_t)NN * DD;          // 1.57M elems
    const size_t SD = (size_t)NS * DD;          // 2.10M elems
    // f32 buffers
    float* feats = (float*)alloc(ND * 4);
    float* hA    = (float*)alloc(ND * 4);
    float* hB    = (float*)alloc(ND * 4);       // becomes re_x
    float* enc01 = (float*)alloc(2 * DD * 4);
    float* cvec  = (float*)alloc(DD * 4);
    float* Q3    = (float*)alloc(192 * DT * 4);
    float* O     = (float*)alloc(192 * DT * 4);
    // f16 activations
    _Float16* xh    = (_Float16*)alloc((size_t)NN * FF * 2);
    _Float16* Z2h   = (_Float16*)alloc(SD * 2);
    _Float16* Z3h   = (_Float16*)alloc(SD * 2);
    _Float16* ENCh  = (_Float16*)alloc((size_t)NS * 2 * DD * 2);
    _Float16* S1h   = (_Float16*)alloc(SD * 2);
    _Float16* STh   = (_Float16*)alloc(SD * 2);
    _Float16* seqh  = (_Float16*)alloc((size_t)BB * SEQ * DT * 2);
    _Float16* KmH   = (_Float16*)alloc((size_t)BB * SEQ * DT * 2);
    _Float16* VmH   = (_Float16*)alloc((size_t)BB * SEQ * DT * 2);
    _Float16* Q3inh = (_Float16*)alloc(192 * DT * 2);
    _Float16* AVh   = (_Float16*)alloc(192 * DT * 2);
    // f16 pre-transposed weights (N x K)
    _Float16* proj_wT = (_Float16*)alloc((size_t)DD * FF * 2);
    _Float16* w2_2T   = (_Float16*)alloc((size_t)DD * DD * 2);
    _Float16* w2_3T   = (_Float16*)alloc((size_t)DD * DD * 2);
    _Float16* W1bT    = (_Float16*)alloc((size_t)DD * 2 * DD * 2);  // 128 x 256
    _Float16* w2sT    = (_Float16*)alloc((size_t)DD * DD * 2);
    _Float16* wqT     = (_Float16*)alloc((size_t)DT * DT * 2);
    _Float16* wkT     = (_Float16*)alloc((size_t)DT * DT * 2);
    _Float16* wvT     = (_Float16*)alloc((size_t)DT * DT * 2);
    _Float16* woT     = (_Float16*)alloc((size_t)DT * DT * 2);

    // ---- one-time weight transpose+convert and activation convert ----
    wtrans_kernel<<<(FF*DD+255)/256, 256, 0, stream>>>(proj_w, proj_wT, FF, DD);
    wtrans_kernel<<<(DD*DD+255)/256, 256, 0, stream>>>(enc_w2 + 2*DD*DD, w2_2T, DD, DD);
    wtrans_kernel<<<(DD*DD+255)/256, 256, 0, stream>>>(enc_w2 + 3*DD*DD, w2_3T, DD, DD);
    wtrans_kernel<<<(2*DD*DD+255)/256, 256, 0, stream>>>(struct_w1 + (size_t)2*DD*DD, W1bT, 2*DD, DD);
    wtrans_kernel<<<(DD*DD+255)/256, 256, 0, stream>>>(struct_w2, w2sT, DD, DD);
    wtrans_kernel<<<(DT*DT+255)/256, 256, 0, stream>>>(wq, wqT, DT, DT);
    wtrans_kernel<<<(DT*DT+255)/256, 256, 0, stream>>>(wk, wkT, DT, DT);
    wtrans_kernel<<<(DT*DT+255)/256, 256, 0, stream>>>(wv, wvT, DT, DT);
    wtrans_kernel<<<(DT*DT+255)/256, 256, 0, stream>>>(wo, woT, DT, DT);
    f32tof16_kernel<<<(int)((NN*(size_t)FF+255)/256), 256, 0, stream>>>(x, xh, NN*FF);

    // 1) feats = x @ proj_w + proj_b            (12288 x 128 x 128)
    gemm_wmma<128, false, float><<<dim3(NN/64, DD/64), 256, 0, stream>>>(
        xh, FF, proj_wT, feats, DD, proj_b);

    // 2) 3 rounds of edge scatter-add, then re_x mix
    const int nzb = (int)((ND + 255) / 256);
    const int npb = (int)(((long long)EE * 32 + 255) / 256);
    zero_kernel<<<nzb, 256, 0, stream>>>(hA, (int)ND);
    prop_kernel<<<npb, 256, 0, stream>>>(feats, hA, adj_vals, adj_rows, adj_cols);
    zero_kernel<<<nzb, 256, 0, stream>>>(hB, (int)ND);
    prop_kernel<<<npb, 256, 0, stream>>>(hA, hB, adj_vals, adj_rows, adj_cols);
    zero_kernel<<<nzb, 256, 0, stream>>>(hA, (int)ND);
    prop_kernel<<<npb, 256, 0, stream>>>(hB, hA, adj_vals, adj_rows, adj_cols);
    mix_kernel<<<nzb, 256, 0, stream>>>(hA, feats, hB, (int)ND);   // hB = re_x

    // 3) heuristic gathers + encoder hidden layers; constant folds
    zbuild_kernel<<<NS, 128, 0, stream>>>(ppr, drnl, srcv, dstv, samp_idx, node_idx,
                                          enc_w1, enc_b1, Z2h, Z3h);
    encconst_kernel<<<1, 128, 0, stream>>>(enc_b1, enc_w2, enc_b2, enc01);
    cvec_kernel<<<1, 128, 0, stream>>>(enc01, struct_w1, struct_b1, cvec);

    // 4) enc2 / enc3 output layers -> ENC = [enc2 | enc3]   (16384x128x128 each)
    gemm_wmma<128, false, _Float16><<<dim3(NS/64, DD/64), 256, 0, stream>>>(
        Z2h, DD, w2_2T, ENCh, 2*DD, enc_b2 + 2*DD);
    gemm_wmma<128, false, _Float16><<<dim3(NS/64, DD/64), 256, 0, stream>>>(
        Z3h, DD, w2_3T, ENCh + DD, 2*DD, enc_b2 + 3*DD);

    // 5) struct layer1: relu(ENC @ W1[256:512] + cvec)      (16384x128x256)
    gemm_wmma<256, true, _Float16><<<dim3(NS/64, DD/64), 256, 0, stream>>>(
        ENCh, 2*DD, W1bT, S1h, DD, cvec);

    // 6) struct layer2                                       (16384x128x128)
    gemm_wmma<128, false, _Float16><<<dim3(NS/64, DD/64), 256, 0, stream>>>(
        S1h, DD, w2sT, STh, DD, struct_b2);

    // 7) assemble token sequence (B x 257 x 256, f16)
    seq_kernel<<<BB * SEQ, 256, 0, stream>>>(STh, hB, node_idx, cls_tok, seqh);

    // 8) full K and V projections                            (16448x256x256)
    gemm_wmma<256, false, _Float16><<<dim3(BB*SEQ/64, DT/64), 256, 0, stream>>>(
        seqh, DT, wkT, KmH, DT, nullptr);
    gemm_wmma<256, false, _Float16><<<dim3(BB*SEQ/64, DT/64), 256, 0, stream>>>(
        seqh, DT, wvT, VmH, DT, nullptr);

    // 9) Q only for the 3 surviving query rows per batch     (192x256x256)
    q3gather_kernel<<<192, 256, 0, stream>>>(seqh, Q3inh);
    gemm_wmma<256, false, float><<<dim3(192/64, DT/64), 256, 0, stream>>>(
        Q3inh, DT, wqT, Q3, DT, nullptr);

    // 10) softmax attention, 3 queries x 257 keys per (batch, head)
    attn_kernel<<<dim3(BB, HH), 256, 0, stream>>>(Q3, KmH, VmH, AVh);

    // 11) output projection                                  (192x256x256)
    gemm_wmma<256, false, float><<<dim3(192/64, DT/64), 256, 0, stream>>>(
        AVh, DT, woT, O, DT, nullptr);

    // 12) scatter to [h_a | h_b | h_cls]
    outscatter_kernel<<<BB, 256, 0, stream>>>(O, (float*)d_out);
}
